// EnhancedSelfAttention_30915174596924
// MI455X (gfx1250) — matrix operand order repaired and checked by reading, not compiled, not run on hardware
//
#include <hip/hip_runtime.h>
#include <hip/hip_bf16.h>

typedef __bf16 bf16;
typedef __attribute__((ext_vector_type(16))) __bf16 v16bf;
typedef __attribute__((ext_vector_type(8)))  __bf16 v8bf;
typedef __attribute__((ext_vector_type(8)))  float  v8f;
typedef __attribute__((ext_vector_type(4)))  unsigned v4u;
typedef __attribute__((ext_vector_type(8)))  unsigned v8u;

#define BATCH 16
#define CHN   256
#define NPIX  1024
#define NH    8
#define HD    32
#define GN_EPS 1e-5f

__device__ __forceinline__ v8f zero8() {
  v8f z = {0.f, 0.f, 0.f, 0.f, 0.f, 0.f, 0.f, 0.f};
  return z;
}

__device__ __forceinline__ v8f wmma_bf16(v16bf a, v16bf b, v8f c) {
  // D = A(16x32 bf16) * B(32x16 bf16) + C(16x16 f32)
  return __builtin_amdgcn_wmma_f32_16x16x32_bf16(false, a, false, b, (short)0, c,
                                                 false, false);
}

// A-fragment: 16x32 (MxK) bf16, src row-major with leading dim ld (elements).
// lane<16 -> row=lane, K {0..7, 16..23}; lane>=16 -> row=lane-16, K {8..15, 24..31}.
__device__ __forceinline__ v16bf load_a_frag(const bf16* base, int ld) {
  int lane = threadIdx.x & 31;
  int half = lane >> 4, r = lane & 15;
  const bf16* p = base + (size_t)r * ld + half * 8;
  v8bf lo = *(const v8bf*)(p);
  v8bf hi = *(const v8bf*)(p + 16);
  v16bf out;
#pragma unroll
  for (int i = 0; i < 8; ++i) { out[i] = lo[i]; out[i + 8] = hi[i]; }
  return out;
}

// B-fragment: 32x16 (KxN) bf16 sourced from B^T stored row-major (N x K, ld elems).
// lane<16 -> col=lane, K {0..7, 16..23}; lane>=16 -> col=lane-16, K {8..15, 24..31}.
__device__ __forceinline__ v16bf load_b_frag(const bf16* base, int ld) {
  int lane = threadIdx.x & 31;
  int half = lane >> 4, r = lane & 15;
  const bf16* p = base + (size_t)r * ld + half * 8;
  v8bf lo = *(const v8bf*)(p);
  v8bf hi = *(const v8bf*)(p + 16);
  v16bf out;
#pragma unroll
  for (int i = 0; i < 8; ++i) { out[i] = lo[i]; out[i + 8] = hi[i]; }
  return out;
}

// ---------------------------------------------------------------- TDM
// DMA a 2D bf16 tile (rows x 256, row stride 256 elems) from global into LDS.
// D# per CDNA5 ISA: group0 = {count/lds_addr/global_addr/type},
// group1 = {data_size, tensor dims, tile dims, strides}. Issued per-wave (EXEC
// ignored) -> caller must gate to a single wave per workgroup.
__device__ __forceinline__ void tdm_load_2d(const bf16* gptr, void* lptr,
                                            unsigned tile_rows,
                                            unsigned tensor_rows) {
  unsigned long long ga = (unsigned long long)(uintptr_t)gptr;
  unsigned lo = (unsigned)(uintptr_t)lptr;  // generic addr low 32b == LDS offset
  v4u g0;
  g0[0] = 1u;                                   // count=1 (valid descriptor)
  g0[1] = lo;                                   // lds_addr (bytes)
  g0[2] = (unsigned)ga;                         // global_addr[31:0]
  g0[3] = ((unsigned)(ga >> 32) & 0x01FFFFFFu)  // global_addr[56:32]
          | (2u << 30);                         // type=2 ("image")
  v8u g1;
  g1[0] = 1u << 16;          // workgroup_mask=0, data_size=1 (2 bytes)
  g1[1] = 256u << 16;        // atomic_barrier=0 | tensor_dim0[15:0]=256
  g1[2] = tensor_rows << 16; // tensor_dim0[31:16]=0 | tensor_dim1[15:0]
  g1[3] = 256u << 16;        // tensor_dim1[31:16]=0 | tile_dim0=256
  g1[4] = tile_rows;         // tile_dim1=rows, tile_dim2=0
  g1[5] = 256u;              // tensor_dim0_stride[31:0]=256
  g1[6] = 0u;                // stride0[47:32]=0 | tensor_dim1_stride lo
  g1[7] = 0u;
  asm volatile("tensor_load_to_lds %0, %1" ::"s"(g0), "s"(g1) : "memory");
}

// ---------------------------------------------------------------- utilities
__global__ void f32_to_bf16_kernel(const float* __restrict__ src,
                                   bf16* __restrict__ dst, int n) {
  int i = blockIdx.x * blockDim.x + threadIdx.x;
  if (i < n) dst[i] = (bf16)src[i];
}

// ------------------------------------------------------ 1. GroupNorm stats
__global__ void gn_stats_kernel(const float* __restrict__ x,
                                float* __restrict__ stats) {
  int b = blockIdx.x;
  const float* xb = x + (size_t)b * CHN * NPIX;
  float s = 0.f, s2 = 0.f;
  for (int i = threadIdx.x; i < CHN * NPIX; i += 256) {
    float v = xb[i];
    s += v;
    s2 += v * v;
  }
  __shared__ float rs[256], rq[256];
  rs[threadIdx.x] = s;
  rq[threadIdx.x] = s2;
  __syncthreads();
  for (int off = 128; off > 0; off >>= 1) {
    if (threadIdx.x < off) {
      rs[threadIdx.x] += rs[threadIdx.x + off];
      rq[threadIdx.x] += rq[threadIdx.x + off];
    }
    __syncthreads();
  }
  if (threadIdx.x == 0) {
    const float inv = 1.f / (float)(CHN * NPIX);
    float mean = rs[0] * inv;
    float var = rq[0] * inv - mean * mean;
    stats[2 * b] = mean;
    stats[2 * b + 1] = rsqrtf(var + GN_EPS);
  }
}

// --------------------------------------- 2. normalize + transpose to (b,n,c)
__global__ void norm_transpose_kernel(const float* __restrict__ x,
                                      const float* __restrict__ stats,
                                      const float* __restrict__ gw,
                                      const float* __restrict__ gb,
                                      bf16* __restrict__ xnT) {
  size_t i = (size_t)blockIdx.x * blockDim.x + threadIdx.x;  // B*C*N total
  int b = (int)(i / (CHN * NPIX));
  int rem = (int)(i % (CHN * NPIX));
  int c = rem / NPIX, n = rem % NPIX;
  float v = (x[i] - stats[2 * b]) * stats[2 * b + 1] * gw[c] + gb[c];
  xnT[((size_t)b * NPIX + n) * CHN + c] = (bf16)v;
}

// ------------------------------------------------- 3. QKV GEMM + scatter
// qkv(o,n) = sum_c W(o,c) * xnT(n,c). Block tile 64o x 64n, full K=256 staged
// in LDS by two TDM descriptors. 8 waves, each 16o x 32n.
__global__ void qkv_kernel(const bf16* __restrict__ w,
                           const float* __restrict__ bias,
                           const bf16* __restrict__ xnT,
                           bf16* __restrict__ qb, bf16* __restrict__ kb,
                           bf16* __restrict__ vT) {
  __shared__ __align__(16) bf16 ldsA[64 * CHN];  // 32 KB weights
  __shared__ __align__(16) bf16 ldsB[64 * CHN];  // 32 KB activations
  int wave = threadIdx.x >> 5;
  int lane = threadIdx.x & 31;
  int blk = blockIdx.x;  // B * 12 * 16
  int b = blk / 192;
  int r = blk % 192;
  int ot = r / 16, nt = r % 16;
  int o_base = ot * 64;
  int n_base = nt * 64;

  if (wave == 0) {
    tdm_load_2d(w + (size_t)o_base * CHN, ldsA, 64, 3 * CHN);
    tdm_load_2d(xnT + ((size_t)b * NPIX + n_base) * CHN, ldsB, 64, NPIX);
    __builtin_amdgcn_s_wait_tensorcnt(0);
  }
  __syncthreads();

  int ow = (wave >> 1) * 16;  // local o row of this wave
  int nw = (wave & 1) * 32;   // local n col of this wave
  v8f acc[2];
#pragma unroll
  for (int t = 0; t < 2; ++t) acc[t] = zero8();

  const bf16* arow = ldsA + (size_t)ow * CHN;
  const bf16* brow = ldsB + (size_t)nw * CHN;
#pragma unroll
  for (int cc = 0; cc < CHN; cc += 32) {
    v16bf af = load_a_frag(arow + cc, CHN);
#pragma unroll
    for (int t = 0; t < 2; ++t) {
      v16bf bf = load_b_frag(brow + (size_t)(t * 16) * CHN + cc, CHN);
      acc[t] = wmma_bf16(af, bf, acc[t]);
    }
  }

  int half = lane >> 4, rl = lane & 15;
  const float scale = 0.17677669529663689f;  // 32^-0.5 folded into Q
#pragma unroll
  for (int t = 0; t < 2; ++t) {
    int n = n_base + nw + t * 16 + rl;
#pragma unroll
    for (int j = 0; j < 8; ++j) {
      int o = o_base + ow + j + 8 * half;
      float val = acc[t][j] + bias[o];
      int m = o >> 8;  // 0=q 1=k 2=v
      int oo = o & 255;
      int h = oo >> 5, d = oo & 31;
      size_t qk_idx = ((size_t)(b * NH + h) * NPIX + n) * HD + d;
      if (m == 0)
        qb[qk_idx] = (bf16)(val * scale);
      else if (m == 1)
        kb[qk_idx] = (bf16)val;
      else  // V stored transposed: (b, h, d, n)
        vT[((size_t)(b * NH + h) * HD + d) * NPIX + n] = (bf16)val;
    }
  }
}

// ------------------------------------------------- 4. flash attention
// one wave handles a 16-row block of one (b,h); K=hd=32 => one WMMA per S tile
__global__ void attn_kernel(const bf16* __restrict__ q,
                            const bf16* __restrict__ k,
                            const bf16* __restrict__ vT,
                            bf16* __restrict__ aoT) {
  __shared__ __align__(16) bf16 pshare[4][16][32];
  int wave = threadIdx.x >> 5, lane = threadIdx.x & 31;
  int half = lane >> 4, rl = lane & 15;
  int rb = blockIdx.x * 4 + wave;  // B*NH*64 row blocks
  int b = rb / (NH * 64);
  int rem = rb % (NH * 64);
  int h = rem / 64;
  int n0 = (rem % 64) * 16;

  const bf16* qh = q + (size_t)(b * NH + h) * NPIX * HD;
  const bf16* kh = k + (size_t)(b * NH + h) * NPIX * HD;
  const bf16* vh = vT + (size_t)(b * NH + h) * HD * NPIX;

  v16bf qf = load_a_frag(qh + (size_t)n0 * HD, HD);

  float m_r[8], l_r[8];
#pragma unroll
  for (int j = 0; j < 8; ++j) { m_r[j] = -3.0e38f; l_r[j] = 0.f; }
  v8f o0 = zero8(), o1 = zero8();
  bf16(*P)[32] = pshare[wave];

  for (int mb = 0; mb < NPIX; mb += 32) {
    // issue all global loads up front: V frags stay in flight during softmax
    v16bf kf0 = load_b_frag(kh + (size_t)mb * HD, HD);
    v16bf kf1 = load_b_frag(kh + (size_t)(mb + 16) * HD, HD);
    v16bf vf0 = load_b_frag(vh + mb, NPIX);                      // d = 0..15
    v16bf vf1 = load_b_frag(vh + (size_t)16 * NPIX + mb, NPIX);  // d = 16..31
    v8f s0 = wmma_bf16(qf, kf0, zero8());
    v8f s1 = wmma_bf16(qf, kf1, zero8());
#pragma unroll
    for (int j = 0; j < 8; ++j) {
      float a0 = s0[j], a1 = s1[j];
      float mx = fmaxf(a0, a1);
#pragma unroll
      for (int off = 1; off < 16; off <<= 1)
        mx = fmaxf(mx, __shfl_xor(mx, off, 32));
      float mnew = fmaxf(m_r[j], mx);
      float alpha = __expf(m_r[j] - mnew);
      float p0 = __expf(a0 - mnew);
      float p1 = __expf(a1 - mnew);
      float ls = p0 + p1;
#pragma unroll
      for (int off = 1; off < 16; off <<= 1) ls += __shfl_xor(ls, off, 32);
      l_r[j] = l_r[j] * alpha + ls;
      m_r[j] = mnew;
      o0[j] *= alpha;
      o1[j] *= alpha;
      P[j + 8 * half][rl] = (bf16)p0;  // D-layout -> (row,col) in LDS
      P[j + 8 * half][16 + rl] = (bf16)p1;
    }
    v16bf pf = load_a_frag(&P[0][0], 32);  // re-shape P into A-fragment
    o0 = wmma_bf16(pf, vf0, o0);
    o1 = wmma_bf16(pf, vf1, o1);
  }

#pragma unroll
  for (int j = 0; j < 8; ++j) {
    float inv = 1.f / l_r[j];
    int n = n0 + j + 8 * half;
    bf16* dst = aoT + ((size_t)b * NPIX + n) * CHN + h * HD;
    dst[rl] = (bf16)(o0[j] * inv);
    dst[16 + rl] = (bf16)(o1[j] * inv);
  }
}

// ------------------------------------------------- 5. proj + bias + residual
__global__ void proj_kernel(const bf16* __restrict__ w,
                            const float* __restrict__ bias,
                            const bf16* __restrict__ aoT,
                            const float* __restrict__ x,
                            float* __restrict__ out) {
  __shared__ __align__(16) bf16 ldsA[64 * CHN];
  __shared__ __align__(16) bf16 ldsB[64 * CHN];
  int wave = threadIdx.x >> 5;
  int lane = threadIdx.x & 31;
  int blk = blockIdx.x;  // B * 4 * 16
  int b = blk / 64;
  int r = blk % 64;
  int ot = r / 16, nt = r % 16;
  int o_base = ot * 64;
  int n_base = nt * 64;

  if (wave == 0) {
    tdm_load_2d(w + (size_t)o_base * CHN, ldsA, 64, CHN);
    tdm_load_2d(aoT + ((size_t)b * NPIX + n_base) * CHN, ldsB, 64, NPIX);
    __builtin_amdgcn_s_wait_tensorcnt(0);
  }
  __syncthreads();

  int ow = (wave >> 1) * 16;
  int nw = (wave & 1) * 32;
  v8f acc[2];
#pragma unroll
  for (int t = 0; t < 2; ++t) acc[t] = zero8();

  const bf16* arow = ldsA + (size_t)ow * CHN;
  const bf16* brow = ldsB + (size_t)nw * CHN;
#pragma unroll
  for (int cc = 0; cc < CHN; cc += 32) {
    v16bf af = load_a_frag(arow + cc, CHN);
#pragma unroll
    for (int t = 0; t < 2; ++t) {
      v16bf bf = load_b_frag(brow + (size_t)(t * 16) * CHN + cc, CHN);
      acc[t] = wmma_bf16(af, bf, acc[t]);
    }
  }

  int half = lane >> 4, rl = lane & 15;
#pragma unroll
  for (int t = 0; t < 2; ++t) {
    int n = n_base + nw + t * 16 + rl;
#pragma unroll
    for (int j = 0; j < 8; ++j) {
      int o = o_base + ow + j + 8 * half;
      size_t idx = ((size_t)b * CHN + o) * NPIX + n;
      out[idx] = acc[t][j] + bias[o] + x[idx];
    }
  }
}

extern "C" void kernel_launch(void* const* d_in, const int* in_sizes, int n_in,
                              void* d_out, int out_size, void* d_ws,
                              size_t ws_size, hipStream_t stream) {
  const float* x      = (const float*)d_in[0];
  const float* gn_w   = (const float*)d_in[1];
  const float* gn_b   = (const float*)d_in[2];
  const float* qkv_w  = (const float*)d_in[3];
  const float* qkv_b  = (const float*)d_in[4];
  const float* proj_w = (const float*)d_in[5];
  const float* proj_b = (const float*)d_in[6];
  float* out = (float*)d_out;

  char* ws = (char*)d_ws;
  const size_t SZ = (size_t)BATCH * NPIX * CHN * sizeof(bf16);  // 8 MB
  float* stats = (float*)ws;  // 16 * 2 f32
  bf16* xnT  = (bf16*)(ws + 256);
  bf16* qbuf = (bf16*)(ws + 256 + 1 * SZ);
  bf16* kbuf = (bf16*)(ws + 256 + 2 * SZ);
  bf16* vbuf = (bf16*)(ws + 256 + 3 * SZ);
  bf16* aoT  = (bf16*)(ws + 256 + 4 * SZ);
  bf16* qkvw = (bf16*)(ws + 256 + 5 * SZ);
  bf16* projw = (bf16*)(ws + 256 + 5 * SZ + (size_t)768 * 256 * sizeof(bf16));

  f32_to_bf16_kernel<<<(768 * 256) / 256, 256, 0, stream>>>(qkv_w, qkvw,
                                                            768 * 256);
  f32_to_bf16_kernel<<<(256 * 256) / 256, 256, 0, stream>>>(proj_w, projw,
                                                            256 * 256);
  gn_stats_kernel<<<BATCH, 256, 0, stream>>>(x, stats);
  norm_transpose_kernel<<<(BATCH * CHN * NPIX) / 256, 256, 0, stream>>>(
      x, stats, gn_w, gn_b, xnT);
  qkv_kernel<<<BATCH * 12 * 16, 256, 0, stream>>>(qkvw, qkv_b, xnT, qbuf, kbuf,
                                                  vbuf);
  attn_kernel<<<(BATCH * NH * 64) / 4, 128, 0, stream>>>(qbuf, kbuf, vbuf,
                                                         aoT);
  proj_kernel<<<BATCH * 4 * 16, 256, 0, stream>>>(projw, proj_b, aoT, x, out);
}
